// _ChannelAttentionModule_19258633356089
// MI455X (gfx1250) — matrix operand order repaired and checked by reading, not compiled, not run on hardware
//
#include <hip/hip_runtime.h>

// ---------------------------------------------------------------------------
// Channel attention: G = F F^T ; P = softmax(rowmax(G) - G) ; out = P F + x
//   B=16, C=512, N=4096 (fp32 in/out)
// Compute-bound -> bf16 WMMA (v_wmma_f32_16x16x32_bf16) with split-bf16
// (hi+lo) compensation for fp32-class accuracy through the razor-sharp
// softmax. Both F and F^T staged in split bf16 so every WMMA fragment is
// filled with contiguous 16B lane loads per the CDNA5 VGPR layouts.
// GEMM waves use 64x64 tiles: 48 WMMA per 32 b128 loads per k-step
// (48 FLOP/B) to stay above the L2 machine-balance point.
// ---------------------------------------------------------------------------

typedef __attribute__((ext_vector_type(16))) __bf16 v16bf;
typedef __attribute__((ext_vector_type(8)))  __bf16 v8bf;
typedef __attribute__((ext_vector_type(4)))  __bf16 v4bf;
typedef __attribute__((ext_vector_type(8)))  float  v8f;
typedef __attribute__((ext_vector_type(4)))  float  v4f;

static constexpr int BATCH = 16;
static constexpr int C     = 512;
static constexpr int N     = 4096;

static constexpr size_t F_ELEMS = (size_t)BATCH * C * N;   // 33,554,432
static constexpr size_t G_ELEMS = (size_t)BATCH * C * C;   //  4,194,304

// ---------------------------------------------------------------------------
// WMMA helpers
// ---------------------------------------------------------------------------
static __device__ __forceinline__ v8f bfwmma(v16bf a, v16bf b, v8f c) {
  // (neg_a, A, neg_b, B, c_mod, C, reuse_a, reuse_b)
  return __builtin_amdgcn_wmma_f32_16x16x32_bf16(false, a, false, b,
                                                 (short)0, c, false, false);
}

// A fragment: 16x32 (MxK). lane<16 : row=lane,    elems 0..7 = K+0..7,  8..15 = K+16..23
//             lane>=16: row=lane-16, elems 0..7 = K+8..15, 8..15 = K+24..31
static __device__ __forceinline__ v16bf load_a_frag(const __bf16* __restrict__ base,
                                                    int row0, int kk, int stride) {
  int lane = threadIdx.x & 31;
  int row  = row0 + (lane & 15);
  int k0   = kk + ((lane & 16) ? 8 : 0);
  const __bf16* p = base + (size_t)row * stride + k0;
  v8bf c0 = *reinterpret_cast<const v8bf*>(p);
  v8bf c1 = *reinterpret_cast<const v8bf*>(p + 16);
  return __builtin_shufflevector(c0, c1, 0, 1, 2, 3, 4, 5, 6, 7,
                                 8, 9, 10, 11, 12, 13, 14, 15);
}

// B fragment: 32x16 (KxN), B[k,n] = src[row0+n, kk+k] (src row-major, K contiguous).
// lane<16 : col=lane,    elems 0..7 = K+0..7,  8..15 = K+16..23 ; lane>=16: col=lane-16, elems 0..7 = K+8..15, 8..15 = K+24..31
static __device__ __forceinline__ v16bf load_b_frag(const __bf16* __restrict__ base,
                                                    int row0, int kk, int stride) {
  int lane = threadIdx.x & 31;
  int row  = row0 + (lane & 15);
  int k0   = kk + ((lane & 16) ? 8 : 0);
  const __bf16* p = base + (size_t)row * stride + k0;
  v8bf c0 = *reinterpret_cast<const v8bf*>(p);
  v8bf c1 = *reinterpret_cast<const v8bf*>(p + 16);
  return __builtin_shufflevector(c0, c1, 0, 1, 2, 3, 4, 5, 6, 7,
                                 8, 9, 10, 11, 12, 13, 14, 15);
}

// ---------------------------------------------------------------------------
// Kernel 1: split x -> bf16 hi/lo, plus LDS-transposed copy (F^T hi/lo)
// ---------------------------------------------------------------------------
__global__ void __launch_bounds__(256)
k_split_transpose(const float* __restrict__ x,
                  __bf16* __restrict__ Fh,  __bf16* __restrict__ Fl,
                  __bf16* __restrict__ Fth, __bf16* __restrict__ Ftl) {
  __shared__ float tile[32][33];
  int bid = blockIdx.x;
  int tn  = bid & 127;          // N-tile (128)
  int tc  = (bid >> 7) & 15;    // C-tile (16)
  int b   = bid >> 11;          // batch
  int tid = threadIdx.x;
  int r   = tid >> 3;           // 0..31
  int c4  = (tid & 7) * 4;      // 0,4,..,28

  size_t xi = ((size_t)(b * C + tc * 32 + r)) * N + tn * 32 + c4;
  v4f v = *reinterpret_cast<const v4f*>(x + xi);
  v4bf h, l;
#pragma unroll
  for (int i = 0; i < 4; ++i) {
    float f = v[i];
    __bf16 hh = (__bf16)f;
    h[i] = hh;
    l[i] = (__bf16)(f - (float)hh);
    tile[r][c4 + i] = f;
  }
  *reinterpret_cast<v4bf*>(Fh + xi) = h;
  *reinterpret_cast<v4bf*>(Fl + xi) = l;

  __syncthreads();

  size_t ti = ((size_t)(b * N + tn * 32 + r)) * C + tc * 32 + c4;
  v4bf th, tl;
#pragma unroll
  for (int i = 0; i < 4; ++i) {
    float f = tile[c4 + i][r];
    __bf16 hh = (__bf16)f;
    th[i] = hh;
    tl[i] = (__bf16)(f - (float)hh);
  }
  *reinterpret_cast<v4bf*>(Fth + ti) = th;
  *reinterpret_cast<v4bf*>(Ftl + ti) = tl;
}

// ---------------------------------------------------------------------------
// Kernel 2: Gram matrix G[b] = F F^T, split-bf16 compensated.
// One wave per 64x64 G tile: 16 accumulators, 48 WMMA / 32 b128 per k-step.
// grid = BATCH*8*8/8 = 128 blocks of 256 threads.
// ---------------------------------------------------------------------------
__global__ void __launch_bounds__(256)
k_gram(const __bf16* __restrict__ Fh, const __bf16* __restrict__ Fl,
       float* __restrict__ G) {
  int wave = threadIdx.x >> 5;
  int gw   = blockIdx.x * 8 + wave;        // 0 .. 1023
  int tj   = gw & 7;
  int ti   = (gw >> 3) & 7;
  int b    = gw >> 6;
  int c0   = ti * 64;
  int d0   = tj * 64;
  const size_t fo = (size_t)b * C * N;
  const __bf16* fh = Fh + fo;
  const __bf16* fl = Fl + fo;

  v8f acc[4][4];
#pragma unroll
  for (int mi = 0; mi < 4; ++mi)
#pragma unroll
    for (int nj = 0; nj < 4; ++nj)
      acc[mi][nj] = (v8f){0.f, 0.f, 0.f, 0.f, 0.f, 0.f, 0.f, 0.f};

  for (int kk = 0; kk < N; kk += 32) {
    v16bf ah[4], al[4], bh[4], bl[4];
#pragma unroll
    for (int mi = 0; mi < 4; ++mi) {
      ah[mi] = load_a_frag(fh, c0 + 16 * mi, kk, N);
      al[mi] = load_a_frag(fl, c0 + 16 * mi, kk, N);
    }
#pragma unroll
    for (int nj = 0; nj < 4; ++nj) {
      bh[nj] = load_b_frag(fh, d0 + 16 * nj, kk, N);
      bl[nj] = load_b_frag(fl, d0 + 16 * nj, kk, N);
    }
#pragma unroll
    for (int mi = 0; mi < 4; ++mi)
#pragma unroll
      for (int nj = 0; nj < 4; ++nj) {
        acc[mi][nj] = bfwmma(ah[mi], bh[nj], acc[mi][nj]);
        acc[mi][nj] = bfwmma(al[mi], bh[nj], acc[mi][nj]);
        acc[mi][nj] = bfwmma(ah[mi], bl[nj], acc[mi][nj]);
      }
  }

  int lane  = threadIdx.x & 31;
  int mhalf = (lane & 16) ? 8 : 0;
  int ncol  = lane & 15;
  float* g = G + (size_t)b * C * C;
#pragma unroll
  for (int mi = 0; mi < 4; ++mi)
#pragma unroll
    for (int nj = 0; nj < 4; ++nj)
#pragma unroll
      for (int v = 0; v < 8; ++v)
        g[(size_t)(c0 + 16 * mi + mhalf + v) * C + d0 + 16 * nj + ncol] =
            acc[mi][nj][v];
}

// ---------------------------------------------------------------------------
// Kernel 3: softmax(rowmax - a) == exp(min_a - a) / sum. One wave per row.
// ---------------------------------------------------------------------------
__global__ void __launch_bounds__(256)
k_softmax(const float* __restrict__ G,
          __bf16* __restrict__ Ph, __bf16* __restrict__ Pl) {
  int wave = threadIdx.x >> 5;
  int lane = threadIdx.x & 31;
  int row  = blockIdx.x * 8 + wave;        // 0 .. BATCH*C-1
  const float* g = G + (size_t)row * C;

  float a[16];
  float mn = 3.402823e38f;
#pragma unroll
  for (int j = 0; j < 16; ++j) {
    a[j] = g[lane + 32 * j];
    mn = fminf(mn, a[j]);
  }
#pragma unroll
  for (int off = 16; off >= 1; off >>= 1)
    mn = fminf(mn, __shfl_xor(mn, off, 32));

  float e[16];
  float s = 0.f;
#pragma unroll
  for (int j = 0; j < 16; ++j) {
    e[j] = __expf(mn - a[j]);
    s += e[j];
  }
#pragma unroll
  for (int off = 16; off >= 1; off >>= 1)
    s += __shfl_xor(s, off, 32);

  float inv = 1.0f / s;
  size_t base = (size_t)row * C;
#pragma unroll
  for (int j = 0; j < 16; ++j) {
    float p = e[j] * inv;
    __bf16 h = (__bf16)p;
    Ph[base + lane + 32 * j] = h;
    Pl[base + lane + 32 * j] = (__bf16)(p - (float)h);
  }
}

// ---------------------------------------------------------------------------
// Kernel 4: out = P @ F + x  (K = C = 512), B operand read from F^T copy.
// One wave per 64x64 out tile. grid = BATCH*8*64/8 = 1024 blocks.
// ---------------------------------------------------------------------------
__global__ void __launch_bounds__(256)
k_apply(const __bf16* __restrict__ Ph, const __bf16* __restrict__ Pl,
        const __bf16* __restrict__ Fth, const __bf16* __restrict__ Ftl,
        const float* __restrict__ x, float* __restrict__ out) {
  int wave = threadIdx.x >> 5;
  int gw   = blockIdx.x * 8 + wave;        // 0 .. 8191
  int tn   = gw & 63;
  int tc   = (gw >> 6) & 7;
  int b    = gw >> 9;
  int c0   = tc * 64;
  int n0   = tn * 64;
  const __bf16* ph  = Ph  + (size_t)b * C * C;
  const __bf16* pl  = Pl  + (size_t)b * C * C;
  const __bf16* fth = Fth + (size_t)b * N * C;
  const __bf16* ftl = Ftl + (size_t)b * N * C;

  v8f acc[4][4];
#pragma unroll
  for (int mi = 0; mi < 4; ++mi)
#pragma unroll
    for (int nj = 0; nj < 4; ++nj)
      acc[mi][nj] = (v8f){0.f, 0.f, 0.f, 0.f, 0.f, 0.f, 0.f, 0.f};

  for (int kk = 0; kk < C; kk += 32) {
    v16bf ah[4], al[4], bh[4], bl[4];
#pragma unroll
    for (int mi = 0; mi < 4; ++mi) {
      ah[mi] = load_a_frag(ph, c0 + 16 * mi, kk, C);
      al[mi] = load_a_frag(pl, c0 + 16 * mi, kk, C);
    }
#pragma unroll
    for (int nj = 0; nj < 4; ++nj) {
      bh[nj] = load_b_frag(fth, n0 + 16 * nj, kk, C);
      bl[nj] = load_b_frag(ftl, n0 + 16 * nj, kk, C);
    }
#pragma unroll
    for (int mi = 0; mi < 4; ++mi)
#pragma unroll
      for (int nj = 0; nj < 4; ++nj) {
        acc[mi][nj] = bfwmma(ah[mi], bh[nj], acc[mi][nj]);
        acc[mi][nj] = bfwmma(al[mi], bh[nj], acc[mi][nj]);
        acc[mi][nj] = bfwmma(ah[mi], bl[nj], acc[mi][nj]);
      }
  }

  int lane  = threadIdx.x & 31;
  int mhalf = (lane & 16) ? 8 : 0;
  int ncol  = lane & 15;
  const size_t xo = (size_t)b * C * N;
#pragma unroll
  for (int mi = 0; mi < 4; ++mi)
#pragma unroll
    for (int nj = 0; nj < 4; ++nj)
#pragma unroll
      for (int v = 0; v < 8; ++v) {
        size_t idx = xo + (size_t)(c0 + 16 * mi + mhalf + v) * N +
                     n0 + 16 * nj + ncol;
        out[idx] = acc[mi][nj][v] + x[idx];
      }
}

// ---------------------------------------------------------------------------
extern "C" void kernel_launch(void* const* d_in, const int* in_sizes, int n_in,
                              void* d_out, int out_size, void* d_ws, size_t ws_size,
                              hipStream_t stream) {
  (void)in_sizes; (void)n_in; (void)out_size; (void)ws_size;
  const float* x = (const float*)d_in[0];
  float* out = (float*)d_out;

  char* ws = (char*)d_ws;
  __bf16* Fh  = (__bf16*)ws;                          // F hi   (64 MiB)
  __bf16* Fl  = Fh + F_ELEMS;                         // F lo   (64 MiB)
  __bf16* Fth = Fl + F_ELEMS;                         // F^T hi (64 MiB)
  __bf16* Ftl = Fth + F_ELEMS;                        // F^T lo (64 MiB)
  float*  G   = (float*)(Ftl + F_ELEMS);              // Gram   (16 MiB)
  __bf16* Ph  = (__bf16*)(G + G_ELEMS);               // attn hi (8 MiB)
  __bf16* Pl  = Ph + G_ELEMS;                         // attn lo (8 MiB)

  k_split_transpose<<<BATCH * (C / 32) * (N / 32), 256, 0, stream>>>(x, Fh, Fl, Fth, Ftl);
  k_gram<<<(BATCH * 8 * 8) / 8, 256, 0, stream>>>(Fh, Fl, G);
  k_softmax<<<(BATCH * C) / 8, 256, 0, stream>>>(G, Ph, Pl);
  k_apply<<<(BATCH * 8 * 64) / 8, 256, 0, stream>>>(Ph, Pl, Fth, Ftl, x, out);
}
